// NequIP_90357521973617
// MI455X (gfx1250) — matrix operand order repaired and checked by reading, not compiled, not run on hardware
//
#include <hip/hip_runtime.h>
#include <stdint.h>

typedef __attribute__((ext_vector_type(16))) __bf16 v16bf;
typedef __attribute__((ext_vector_type(8)))  float  v8f;

struct alignas(16) U128 { unsigned int x, y, z, w; };
union V16U { U128 q[2]; v16bf v; };

__device__ __forceinline__ uint16_t f2bf(float f) {
  union { float f; unsigned int u; } c; c.f = f;
  unsigned int u = c.u;
  unsigned int r = u + 0x7FFFu + ((u >> 16) & 1u);   // round-to-nearest-even
  return (uint16_t)(r >> 16);
}

// ---------------------------------------------------------------------------
// Prep: repack W1 (18x256, K padded to 32) and W2 (256x1024) as bf16 tiles in
// the per-lane k order this file's WMMA loads use (contiguous, same as A).
//   B-layout (built here): lanes 0-15 hold K = kc*32 + 0..15,
//                          lanes 16-31 hold K = kc*32 + 16..31, N = lane&15.
// Each lane's 16 bf16 values are stored contiguously (32B) -> 2x b128 loads.
// ---------------------------------------------------------------------------
__global__ void pack_weights(const float* __restrict__ W1,
                             const float* __restrict__ W2,
                             uint16_t* __restrict__ w1p,
                             uint16_t* __restrict__ w2p) {
  int gid = blockIdx.x * 256 + threadIdx.x;
  if (gid < 64 * 8 * 32 * 16) {                 // W2: 64 N-tiles x 8 K-chunks
    int e    = gid & 15;
    int lane = (gid >> 4) & 31;
    int kc   = (gid >> 9) & 7;
    int nt   = gid >> 12;
    int K    = kc * 32 + ((lane < 16) ? 0 : 16) + e;
    int col  = nt * 16 + (lane & 15);
    w2p[gid] = f2bf(W2[K * 1024 + col]);
  }
  if (gid < 16 * 32 * 16) {                     // W1: 16 N-tiles x 1 K-chunk
    int e    = gid & 15;
    int lane = (gid >> 4) & 31;
    int nt   = gid >> 9;
    int K    = ((lane < 16) ? 0 : 16) + e;
    int col  = nt * 16 + (lane & 15);
    w1p[gid] = (K < 18) ? f2bf(W1[K * 256 + col]) : (uint16_t)0;
  }
}

// ---------------------------------------------------------------------------
// Fused main kernel: 64 edges / workgroup, 192 threads = 6 wave32.
// ---------------------------------------------------------------------------
__global__ __launch_bounds__(192) void nequip_fused(
    const float* __restrict__ node_features,
    const float* __restrict__ edge_attr,
    const float* __restrict__ edge_embedding,
    const float* __restrict__ b1,
    const float* __restrict__ b2,
    const int*   __restrict__ edge_src,
    const int*   __restrict__ edge_dst,
    const uint16_t* __restrict__ w1p,
    const uint16_t* __restrict__ w2p,
    float* __restrict__ out)
{
  __shared__ uint16_t sh_h[64 * 256];     // h in bf16, A-layout rows (32 KB)
  __shared__ float    sh_c[6 * 64 * 16];  // 6 contraction vectors per edge
  __shared__ float    sh_res[6 * 64 * 16];// contraction results
  __shared__ uint16_t sh_embA[64 * 32];   // emb bf16, K padded to 32
  __shared__ float    sh_y[64 * 4];       // edge_attr
  __shared__ float    sh_b2[1024];

  const int tid  = threadIdx.x;
  const int e0   = blockIdx.x * 64;
  const int wave = tid >> 5;
  const int lane = tid & 31;
  const int l16  = lane & 15;
  const int hi   = lane >> 4;             // 0: lanes 0-15, 1: lanes 16-31

  // ---- phase 0: stage per-edge data ----
  for (int idx = tid; idx < 256; idx += 192)
    sh_y[idx] = edge_attr[e0 * 4 + idx];
  for (int idx = tid; idx < 1024; idx += 192)
    sh_b2[idx] = b2[idx];
  for (int idx = tid; idx < 64 * 32; idx += 192) {
    int r = idx >> 5, k = idx & 31;
    sh_embA[idx] = (k < 18) ? f2bf(edge_embedding[(e0 + r) * 18 + k])
                            : (uint16_t)0;
  }
  __syncthreads();

  for (int idx = tid; idx < 1024; idx += 192) {   // (edge, i) pairs
    int e = idx >> 4, i = idx & 15;
    int src = edge_src[e0 + e];
    const float* x = node_features + src * 64;
    float s  = x[i];
    float v0 = x[16 + i * 3 + 0];
    float v1 = x[16 + i * 3 + 1];
    float v2 = x[16 + i * 3 + 2];
    float y0 = sh_y[e * 4 + 0];
    float ya = sh_y[e * 4 + 1];
    float yb = sh_y[e * 4 + 2];
    float yc = sh_y[e * 4 + 3];
    sh_c[(0 * 64 + e) * 16 + i] = y0 * s;                                   // ss
    sh_c[(1 * 64 + e) * 16 + i] = 0.57735026919f * (ya*v0 + yb*v1 + yc*v2); // vv
    sh_c[(2 * 64 + e) * 16 + i] = s;                                        // sv
    sh_c[(3 * 64 + e) * 16 + i] = v0;                                       // vs,a=0
    sh_c[(4 * 64 + e) * 16 + i] = v1;                                       // vs,a=1
    sh_c[(5 * 64 + e) * 16 + i] = v2;                                       // vs,a=2
  }
  __syncthreads();

  // ---- phase 1: h = silu(emb @ W1 + b1), one WMMA per 16x16 tile ----
  for (int t = wave; t < 64; t += 6) {
    int m = t >> 4, nt = t & 15;
    const uint16_t* ap = sh_embA + (m * 16 + l16) * 32 + hi * 16;
    V16U av; av.q[0] = *(const U128*)(ap); av.q[1] = *(const U128*)(ap + 8);
    const U128* bp = (const U128*)(w1p) + (nt * 32 + lane) * 2;
    V16U bv; bv.q[0] = bp[0]; bv.q[1] = bp[1];
    v8f cz = {};
    v8f h8 = __builtin_amdgcn_wmma_f32_16x16x32_bf16(
        false, av.v, false, bv.v, (short)0, cz, false, false);
    int col = nt * 16 + l16;
    float bias = b1[col];
#pragma unroll
    for (int r = 0; r < 8; ++r) {
      float xv = h8[r] + bias;
      // silu via hw reciprocal (1-ulp v_rcp_f32): result is rounded to bf16
      // anyway, so the IEEE-divide sequence is pure waste.
      float hv = xv * __builtin_amdgcn_rcpf(1.0f + __expf(-xv));
      int row = m * 16 + r + hi * 8;
      sh_h[row * 256 + col] = f2bf(hv);
    }
  }
  __syncthreads();

  // ---- phase 2: GEMM2 + i-contraction. wave = contraction type t ----
  {
    const int t = wave;                 // 0:ss 1:vv 2:sv 3..5:vs_a
    v8f zero = {};
    v8f acc0 = zero, acc1 = zero, acc2 = zero, acc3 = zero;
    for (int i = 0; i < 16; ++i) {
      v8f c0 = zero, c1 = zero, c2 = zero, c3 = zero;
      int ntB = t * 16 + i;             // column tile = (t,i) slab, all j
#pragma unroll
      for (int kc = 0; kc < 8; ++kc) {
        const U128* bp = (const U128*)(w2p) + ((ntB * 8 + kc) * 32 + lane) * 2;
        V16U bv; bv.q[0] = bp[0]; bv.q[1] = bp[1];
        const uint16_t* base = sh_h + l16 * 256 + kc * 32 + hi * 16;
        V16U a0; a0.q[0] = *(const U128*)(base);
                 a0.q[1] = *(const U128*)(base + 8);
        V16U a1; a1.q[0] = *(const U128*)(base + 16 * 256);
                 a1.q[1] = *(const U128*)(base + 16 * 256 + 8);
        V16U a2; a2.q[0] = *(const U128*)(base + 32 * 256);
                 a2.q[1] = *(const U128*)(base + 32 * 256 + 8);
        V16U a3; a3.q[0] = *(const U128*)(base + 48 * 256);
                 a3.q[1] = *(const U128*)(base + 48 * 256 + 8);
        c0 = __builtin_amdgcn_wmma_f32_16x16x32_bf16(false, a0.v, false, bv.v, (short)0, c0, false, false);
        c1 = __builtin_amdgcn_wmma_f32_16x16x32_bf16(false, a1.v, false, bv.v, (short)0, c1, false, false);
        c2 = __builtin_amdgcn_wmma_f32_16x16x32_bf16(false, a2.v, false, bv.v, (short)0, c2, false, false);
        c3 = __builtin_amdgcn_wmma_f32_16x16x32_bf16(false, a3.v, false, bv.v, (short)0, c3, false, false);
      }
      // fold Sum_i c_t[e,i] * tile  (f32, per-row scalar from LDS)
#pragma unroll
      for (int r = 0; r < 8; ++r) {
        int er = r + hi * 8;
        acc0[r] = fmaf(sh_c[(t * 64 +  0 + er) * 16 + i], c0[r], acc0[r]);
        acc1[r] = fmaf(sh_c[(t * 64 + 16 + er) * 16 + i], c1[r], acc1[r]);
        acc2[r] = fmaf(sh_c[(t * 64 + 32 + er) * 16 + i], c2[r], acc2[r]);
        acc3[r] = fmaf(sh_c[(t * 64 + 48 + er) * 16 + i], c3[r], acc3[r]);
      }
    }
#pragma unroll
    for (int r = 0; r < 8; ++r) {
      int er = r + hi * 8;
      sh_res[(t * 64 +  0 + er) * 16 + l16] = acc0[r];
      sh_res[(t * 64 + 16 + er) * 16 + l16] = acc1[r];
      sh_res[(t * 64 + 32 + er) * 16 + l16] = acc2[r];
      sh_res[(t * 64 + 48 + er) * 16 + l16] = acc3[r];
    }
  }
  __syncthreads();

  // ---- phase 3: combine (+b2 terms), scale, scatter-add ----
  // c_s = sqrt(1/32); c_v*inv_sqrt3 = sqrt(3/32)/sqrt(3) = sqrt(1/32);
  // both folded with the final 1/sqrt(E/N) = 0.25:
  // 0.17677669529 * 0.25 = 0.04419417382.
  const float kScale = 0.04419417382f;
  for (int it = tid; it < 4096; it += 192) {
    int e = it >> 6, f = it & 63;
    float y0 = sh_y[e * 4 + 0];
    float val;
    if (f < 16) {
      int j = f;
      float sum = sh_res[(0 * 64 + e) * 16 + j] + sh_res[(1 * 64 + e) * 16 + j];
      float bsum = 0.0f;
#pragma unroll
      for (int i = 0; i < 16; ++i) {
        bsum += sh_c[(0 * 64 + e) * 16 + i] * sh_b2[i * 16 + j]
              + sh_c[(1 * 64 + e) * 16 + i] * sh_b2[256 + i * 16 + j];
      }
      val = kScale * (sum + bsum);
    } else {
      int g = f - 16;
      int j = g / 3, a = g % 3;
      float ya = sh_y[e * 4 + 1 + a];
      float sv = sh_res[(2 * 64 + e) * 16 + j];
      float vs = sh_res[((3 + a) * 64 + e) * 16 + j];
      float bsv = 0.0f, bvs = 0.0f;
#pragma unroll
      for (int i = 0; i < 16; ++i) {
        bsv += sh_c[(2 * 64 + e) * 16 + i] * sh_b2[512 + i * 16 + j];
        bvs += sh_c[((3 + a) * 64 + e) * 16 + i] * sh_b2[768 + i * 16 + j];
      }
      val = kScale * ((sv + bsv) * ya + y0 * (vs + bvs));
    }
    int dst = edge_dst[e0 + e];
    atomicAdd(out + dst * 64 + f, val);
  }
}

extern "C" void kernel_launch(void* const* d_in, const int* in_sizes, int n_in,
                              void* d_out, int out_size, void* d_ws, size_t ws_size,
                              hipStream_t stream) {
  const float* node_features  = (const float*)d_in[0];
  const float* edge_attr      = (const float*)d_in[1];
  const float* edge_embedding = (const float*)d_in[2];
  const float* W1             = (const float*)d_in[3];
  const float* b1             = (const float*)d_in[4];
  const float* W2             = (const float*)d_in[5];
  const float* b2             = (const float*)d_in[6];
  const int*   edge_src       = (const int*)d_in[7];
  const int*   edge_dst       = (const int*)d_in[8];
  float* out = (float*)d_out;

  uint16_t* w2p = (uint16_t*)d_ws;                                  // 512 KB
  uint16_t* w1p = (uint16_t*)((char*)d_ws + (size_t)64*8*32*16*2);  // +16 KB

  hipMemsetAsync(d_out, 0, (size_t)out_size * sizeof(float), stream);
  pack_weights<<<1024, 256, 0, stream>>>(W1, W2, w1p, w2p);
  nequip_fused<<<2500, 192, 0, stream>>>(node_features, edge_attr, edge_embedding,
                                         b1, b2, edge_src, edge_dst,
                                         w1p, w2p, out);
}
